// LinformerMultiHeadSelfAttentionProjectLayerwise_13280038879465
// MI455X (gfx1250) — matrix operand order repaired and checked by reading, not compiled, not run on hardware
//
#include <hip/hip_runtime.h>
#include <hip/hip_bf16.h>
#include <stdint.h>

#define SEQ_LEN  4096
#define FEAT_DIM 1024
#define NUM_HEAD 16
#define HEAD_DIM 64
#define PROJ_DIM 256
#define BATCH    4

typedef __attribute__((ext_vector_type(16))) __bf16       v16bf;
typedef __attribute__((ext_vector_type(8)))  float        v8f;
typedef __attribute__((ext_vector_type(4)))  unsigned int u32x4;
typedef __attribute__((ext_vector_type(8)))  int          i32x8;
typedef __attribute__((ext_vector_type(4)))  int          i32x4;

union Frag { v16bf v; uint4 q[2]; };

__device__ __forceinline__ v8f wmma_bf16(const Frag& a, const Frag& b, v8f c) {
  return __builtin_amdgcn_wmma_f32_16x16x32_bf16(false, a.v, false, b.v,
                                                 (short)0, c, false, false);
}

// ---------------------------------------------------------------------------
// f32 -> bf16 elementwise convert
// ---------------------------------------------------------------------------
__global__ void cvt_f32_bf16(const float* __restrict__ in, __bf16* __restrict__ out,
                             long long n) {
  long long i = (long long)blockIdx.x * blockDim.x + threadIdx.x;
  long long stride = (long long)gridDim.x * blockDim.x;
  for (; i < n; i += stride) out[i] = (__bf16)in[i];
}

// ---------------------------------------------------------------------------
// f32 [R,C] -> bf16 [C,R] transpose+convert via LDS tile (R,C multiples of 32)
// ---------------------------------------------------------------------------
__global__ void transpose_cvt(const float* __restrict__ in, __bf16* __restrict__ out,
                              int R, int C, long long sIn, long long sOut) {
  __shared__ float t[32][33];
  in  += (long long)blockIdx.z * sIn;
  out += (long long)blockIdx.z * sOut;
  int c0 = blockIdx.x * 32, r0 = blockIdx.y * 32;
  for (int j = threadIdx.y; j < 32; j += 8)
    t[j][threadIdx.x] = in[(long long)(r0 + j) * C + c0 + threadIdx.x];
  __syncthreads();
  for (int j = threadIdx.y; j < 32; j += 8)
    out[(long long)(c0 + j) * R + r0 + threadIdx.x] = (__bf16)t[threadIdx.x][j];
}

// ---------------------------------------------------------------------------
// Generic batched bf16 GEMM: C[m,n] = sum_k A[m,k] * Bt[n,k]
// A row-major [M,K] lda, Bt row-major [N,K] ldb, C [M,N] ldc (f32 or bf16).
// Block = 128 threads = 4 waves; wave computes a 16x64 strip of C.
// grid = (N/64, M/64, batches). Requires M,N % 64 == 0, K % 32 == 0.
// ---------------------------------------------------------------------------
template<bool OUT_BF16>
__global__ void __launch_bounds__(128)
gemm_bf16(const __bf16* __restrict__ A, const __bf16* __restrict__ Bt,
          void* __restrict__ Cv, int K, int lda, int ldb, int ldc,
          long long sA, long long sB, long long sC)
{
  A  += (long long)blockIdx.z * sA;
  Bt += (long long)blockIdx.z * sB;
  const int wave = threadIdx.x >> 5;
  const int lane = threadIdx.x & 31;
  const int row  = lane & 15;
  const int half = lane >> 4;
  const long long m0 = (long long)blockIdx.y * 64 + wave * 16;
  const long long n0 = (long long)blockIdx.x * 64;

  const __bf16* ap = A + (m0 + row) * (long long)lda + half * 8;
  v8f z = {0.f,0.f,0.f,0.f,0.f,0.f,0.f,0.f};
  v8f acc[4] = { z, z, z, z };

  for (int k = 0; k < K; k += 32) {
    __builtin_prefetch(ap + k + 128, 0, 1);   // global_prefetch_b8 on A stream
    Frag a;
    a.q[0] = *(const uint4*)(ap + k);         // K = k + 8*half   .. +7
    a.q[1] = *(const uint4*)(ap + k + 16);    // K = k + 16+8*half.. +7
#pragma unroll
    for (int nt = 0; nt < 4; ++nt) {
      const __bf16* bp = Bt + (n0 + nt * 16 + row) * (long long)ldb + k + half * 8;
      Frag b;
      b.q[0] = *(const uint4*)(bp);           // K = k + 8*half .. +7
      b.q[1] = *(const uint4*)(bp + 16);      // K = k + 16+8*half .. +7
      acc[nt] = wmma_bf16(a, b, acc[nt]);
    }
  }

#pragma unroll
  for (int nt = 0; nt < 4; ++nt) {
#pragma unroll
    for (int i = 0; i < 8; ++i) {
      long long m = m0 + i + half * 8;        // f32 C layout: VGPR i -> row i (+8 hi half)
      long long n = n0 + nt * 16 + row;
      float val = acc[nt][i];
      if (OUT_BF16)
        ((__bf16*)Cv)[(long long)blockIdx.z * sC + m * ldc + n] = (__bf16)val;
      else
        ((float*)Cv)[(long long)blockIdx.z * sC + m * ldc + n] = val;
    }
  }
}

// ---------------------------------------------------------------------------
// TDM descriptor builders (D# group 0 / group 1), per CDNA5 ISA.
// data_size = 8 bytes (code 3); all dims/strides in 8-byte units.
// ---------------------------------------------------------------------------
__device__ __forceinline__ u32x4 make_g0(uint32_t lds_off, unsigned long long gaddr) {
  u32x4 g;
  g[0] = 1u;                                                  // count=1, user desc
  g[1] = lds_off;                                             // lds_addr (bytes)
  g[2] = (uint32_t)(gaddr & 0xffffffffu);                     // global_addr[31:0]
  g[3] = (uint32_t)((gaddr >> 32) & 0x1ffffffu) | (2u << 30); // ga[56:32] | type=2
  return g;
}
__device__ __forceinline__ i32x8 make_g1(uint32_t tdim0, uint32_t tdim1,
                                         uint32_t tile0, uint32_t tile1,
                                         unsigned long long stride0) {
  i32x8 g;
  g[0] = (int)(3u << 16);                                     // mask=0, data_size=8B
  g[1] = (int)((tdim0 & 0xffffu) << 16);                      // tensor_dim0[15:0]
  g[2] = (int)((tdim0 >> 16) | ((tdim1 & 0xffffu) << 16));    // td0[31:16] | td1[15:0]
  g[3] = (int)((tdim1 >> 16) | ((tile0 & 0xffffu) << 16));    // td1[31:16] | tile_dim0
  g[4] = (int)(tile1 & 0xffffu);                              // tile_dim1 (tile_dim2=0)
  g[5] = (int)(stride0 & 0xffffffffu);                        // tensor_dim0_stride lo
  g[6] = (int)((stride0 >> 32) & 0xffffu);                    // stride0 hi (dim1_stride=0)
  g[7] = 0;
  return g;
}

// ---------------------------------------------------------------------------
// Fused Linformer attention for one (b, h, 64-row tile).
// q:[B,SEQ,FEAT] bf16; kproj:[B,PROJ,FEAT] bf16; vprojT:[B,FEAT,PROJ] bf16;
// outb:[B,SEQ,FEAT] bf16. Block = 128 threads (4 waves, 16 rows each).
// Dynamic LDS = 64KB: [0,32K) k_proj tile (reused for P), [32K,64K) vpT tile.
// ---------------------------------------------------------------------------
__global__ void __launch_bounds__(128)
linformer_attn(const __bf16* __restrict__ q, const __bf16* __restrict__ kproj,
               const __bf16* __restrict__ vprojT, __bf16* __restrict__ outb)
{
  extern __shared__ __align__(16) char smem[];
  __bf16* kp_s = (__bf16*)smem;                               // [256][64]
  __bf16* vp_s = (__bf16*)(smem + PROJ_DIM * HEAD_DIM * 2);   // [64][256]
  __bf16* p_s  = (__bf16*)smem;                               // alias: [64][256] probs

  const int b = blockIdx.z, h = blockIdx.y;
  const int wave = threadIdx.x >> 5, lane = threadIdx.x & 31;
  const int row = lane & 15, half = lane >> 4;
  const long long row0 = (long long)blockIdx.x * 64 + wave * 16;

  // ---- TDM: stage K/V head tiles into LDS (wave 0 issues, all sync) ----
  if (wave == 0) {
    const __bf16* kb = kproj  + ((long long)b * PROJ_DIM) * FEAT_DIM + h * HEAD_DIM;
    const __bf16* vb = vprojT + ((long long)b * FEAT_DIM + h * HEAD_DIM) * PROJ_DIM;
    i32x4 z4 = {0, 0, 0, 0};
    i32x8 z8 = {0, 0, 0, 0, 0, 0, 0, 0};
    // k_proj tile: tensor [256 rows x 1024 el], tile 64 el wide x 256 rows
    __builtin_amdgcn_tensor_load_to_lds(
        make_g0((uint32_t)(uintptr_t)kp_s, (unsigned long long)(uintptr_t)kb),
        make_g1(FEAT_DIM * 2 / 8, PROJ_DIM, HEAD_DIM * 2 / 8, PROJ_DIM, FEAT_DIM * 2 / 8),
        z4, z4, z8, 0);
    // vpT tile: tensor [1024 rows x 256 el], tile 256 el wide x 64 rows
    __builtin_amdgcn_tensor_load_to_lds(
        make_g0((uint32_t)(uintptr_t)vp_s, (unsigned long long)(uintptr_t)vb),
        make_g1(PROJ_DIM * 2 / 8, FEAT_DIM, PROJ_DIM * 2 / 8, HEAD_DIM, PROJ_DIM * 2 / 8),
        z4, z4, z8, 0);
    __builtin_amdgcn_s_wait_tensorcnt(0);
  }
  __syncthreads();

  // ---- Q fragments: 16 rows x K=64 (two k-steps of 32) ----
  const __bf16* qp = q + ((long long)b * SEQ_LEN + row0 + row) * FEAT_DIM
                       + h * HEAD_DIM + half * 8;
  Frag qa[2];
#pragma unroll
  for (int kk = 0; kk < 2; ++kk) {
    qa[kk].q[0] = *(const uint4*)(qp + kk * 32);
    qa[kk].q[1] = *(const uint4*)(qp + kk * 32 + 16);
  }

  v8f z = {0.f,0.f,0.f,0.f,0.f,0.f,0.f,0.f};
  v8f acc[16];
#pragma unroll
  for (int t = 0; t < 16; ++t) acc[t] = z;

  // ---- scores[16 x 256] = q @ kp^T ----
#pragma unroll
  for (int t = 0; t < 16; ++t) {
#pragma unroll
    for (int kk = 0; kk < 2; ++kk) {
      const __bf16* bp = kp_s + (t * 16 + row) * HEAD_DIM + kk * 32 + half * 8;
      Frag bf;
      bf.q[0] = *(const uint4*)bp;
      bf.q[1] = *(const uint4*)(bp + 16);
      acc[t] = wmma_bf16(qa[kk], bf, acc[t]);
    }
  }

  // ---- softmax over 256 (scale 1/sqrt(64)); C-layout row m lives in VGPR m%8,
  //      lanes split at 16, so reduce over 16 tiles in-lane + 4 shfl_xor steps ----
  const float scale = 0.125f;
  float rmax[8], rinv[8];
#pragma unroll
  for (int i = 0; i < 8; ++i) {
    float m = -3.0e38f;
#pragma unroll
    for (int t = 0; t < 16; ++t) m = fmaxf(m, acc[t][i]);
#pragma unroll
    for (int s = 1; s < 16; s <<= 1) m = fmaxf(m, __shfl_xor(m, s, 32));
    rmax[i] = m;
  }
#pragma unroll
  for (int i = 0; i < 8; ++i) {
    float s = 0.f;
#pragma unroll
    for (int t = 0; t < 16; ++t) {
      float p = __expf((acc[t][i] - rmax[i]) * scale);
      acc[t][i] = p;
      s += p;
    }
#pragma unroll
    for (int sh = 1; sh < 16; sh <<= 1) s += __shfl_xor(s, sh, 32);
    rinv[i] = 1.0f / s;
  }

  __syncthreads();   // everyone done reading kp_s -> safe to reuse as P staging

  // ---- re-lay out P through LDS into A-fragment order (unnormalized) ----
  __bf16* pw = p_s + wave * 16 * PROJ_DIM;
#pragma unroll
  for (int t = 0; t < 16; ++t)
#pragma unroll
    for (int i = 0; i < 8; ++i)
      pw[(i + half * 8) * PROJ_DIM + t * 16 + row] = (__bf16)acc[t][i];
  __syncthreads();

  // ---- out[16 x 64] = P @ vpT^T (K = 256, 8 k-steps) ----
  v8f oacc[4] = { z, z, z, z };
#pragma unroll
  for (int kk = 0; kk < 8; ++kk) {
    Frag af;
    const __bf16* ap = pw + row * PROJ_DIM + kk * 32 + half * 8;
    af.q[0] = *(const uint4*)ap;
    af.q[1] = *(const uint4*)(ap + 16);
#pragma unroll
    for (int td = 0; td < 4; ++td) {
      const __bf16* bp = vp_s + (td * 16 + row) * PROJ_DIM + kk * 32 + half * 8;
      Frag bf;
      bf.q[0] = *(const uint4*)bp;
      bf.q[1] = *(const uint4*)(bp + 16);
      oacc[td] = wmma_bf16(af, bf, oacc[td]);
    }
  }

  // ---- normalize by row sum and store bf16 ----
  __bf16* op = outb + ((long long)b * SEQ_LEN + row0) * FEAT_DIM + h * HEAD_DIM;
#pragma unroll
  for (int td = 0; td < 4; ++td)
#pragma unroll
    for (int i = 0; i < 8; ++i)
      op[(long long)(i + half * 8) * FEAT_DIM + td * 16 + row] =
          (__bf16)(oacc[td][i] * rinv[i]);
}

// ---------------------------------------------------------------------------
// Host orchestration
// ---------------------------------------------------------------------------
extern "C" void kernel_launch(void* const* d_in, const int* in_sizes, int n_in,
                              void* d_out, int out_size, void* d_ws, size_t ws_size,
                              hipStream_t stream) {
  (void)in_sizes; (void)n_in; (void)out_size; (void)ws_size;
  const float* x    = (const float*)d_in[0];
  const float* proj = (const float*)d_in[1];
  const float* Wq   = (const float*)d_in[2];
  const float* Wk   = (const float*)d_in[3];
  const float* Wv   = (const float*)d_in[4];
  const float* Wo   = (const float*)d_in[5];
  float* out = (float*)d_out;

  const long long NTOK = (long long)BATCH * SEQ_LEN;          // 16384
  const long long NX   = NTOK * FEAT_DIM;                     // 16.7M
  const long long NW   = (long long)FEAT_DIM * FEAT_DIM;      // 1M
  const long long NPX  = (long long)BATCH * PROJ_DIM * FEAT_DIM;

  char* ws = (char*)d_ws;
  size_t off = 0;
  auto alloc = [&](size_t bytes) {
    void* p = ws + off;
    off = (off + bytes + 255) & ~(size_t)255;
    return p;
  };
  __bf16* xb  = (__bf16*)alloc(NX * 2);                       // bf16(x)  [B*N, F]
  __bf16* xT  = (__bf16*)alloc(NX * 2);                       // x^T per batch [F, N]
  __bf16* Pt  = (__bf16*)alloc((size_t)PROJ_DIM * SEQ_LEN * 2);
  __bf16* Wqb = (__bf16*)alloc(NW * 2);
  __bf16* Wkb = (__bf16*)alloc(NW * 2);
  __bf16* Wvb = (__bf16*)alloc(NW * 2);
  __bf16* Wob = (__bf16*)alloc(NW * 2);
  __bf16* qb  = (__bf16*)alloc(NX * 2);                       // Q        [B*N, F]
  __bf16* xp  = (__bf16*)alloc(NPX * 2);                      // P^T x    [B, 256, F]
  __bf16* kp  = (__bf16*)alloc(NPX * 2);                      // k_proj   [B, 256, F]
  __bf16* vpT = (__bf16*)alloc(NPX * 2);                      // v_proj^T [B, F, 256]
  __bf16* ao  = (__bf16*)alloc(NX * 2);                       // attn out [B*N, F]

  // 1) convert inputs to bf16 (+ transposed copies where contraction is over rows)
  cvt_f32_bf16<<<2048, 256, 0, stream>>>(x, xb, NX);
  cvt_f32_bf16<<<512, 256, 0, stream>>>(Wq, Wqb, NW);
  cvt_f32_bf16<<<512, 256, 0, stream>>>(Wk, Wkb, NW);
  cvt_f32_bf16<<<512, 256, 0, stream>>>(Wv, Wvb, NW);
  cvt_f32_bf16<<<512, 256, 0, stream>>>(Wo, Wob, NW);
  transpose_cvt<<<dim3(FEAT_DIM / 32, SEQ_LEN / 32, BATCH), dim3(32, 8), 0, stream>>>(
      x, xT, SEQ_LEN, FEAT_DIM, (long long)SEQ_LEN * FEAT_DIM, (long long)SEQ_LEN * FEAT_DIM);
  transpose_cvt<<<dim3(PROJ_DIM / 32, SEQ_LEN / 32, 1), dim3(32, 8), 0, stream>>>(
      proj, Pt, SEQ_LEN, PROJ_DIM, 0, 0);

  // 2) Q = x @ Wq^T                            [16384,1024] x [1024,1024]
  gemm_bf16<true><<<dim3(FEAT_DIM / 64, NTOK / 64, 1), 128, 0, stream>>>(
      xb, Wqb, qb, FEAT_DIM, FEAT_DIM, FEAT_DIM, FEAT_DIM, 0, 0, 0);

  // 3) xp[b] = P^T @ x[b]  (A=Pt [256,4096], Bt=xT[b] [1024,4096])
  gemm_bf16<true><<<dim3(FEAT_DIM / 64, PROJ_DIM / 64, BATCH), 128, 0, stream>>>(
      Pt, xT, xp, SEQ_LEN, SEQ_LEN, SEQ_LEN, FEAT_DIM,
      0, (long long)FEAT_DIM * SEQ_LEN, (long long)PROJ_DIM * FEAT_DIM);

  // 4) k_proj[b] = xp[b] @ Wk^T                [256,1024] x [1024,1024]
  gemm_bf16<true><<<dim3(FEAT_DIM / 64, PROJ_DIM / 64, BATCH), 128, 0, stream>>>(
      xp, Wkb, kp, FEAT_DIM, FEAT_DIM, FEAT_DIM, FEAT_DIM,
      (long long)PROJ_DIM * FEAT_DIM, 0, (long long)PROJ_DIM * FEAT_DIM);

  // 5) v_proj^T[b] = Wv @ xp[b]^T  (A=Wv [1024,1024], Bt=xp[b] [256,1024])
  gemm_bf16<true><<<dim3(PROJ_DIM / 64, FEAT_DIM / 64, BATCH), 128, 0, stream>>>(
      Wvb, xp, vpT, FEAT_DIM, FEAT_DIM, FEAT_DIM, PROJ_DIM,
      0, (long long)PROJ_DIM * FEAT_DIM, (long long)FEAT_DIM * PROJ_DIM);

  // 6) fused attention (TDM-staged K/V tiles, WMMA scores, softmax, WMMA P@V)
  linformer_attn<<<dim3(SEQ_LEN / 64, NUM_HEAD, BATCH), 128, 65536, stream>>>(
      qb, kp, vpT, ao);

  // 7) out = attn_out @ Wo^T  (f32 output)
  gemm_bf16<false><<<dim3(FEAT_DIM / 64, NTOK / 64, 1), 128, 0, stream>>>(
      ao, Wob, out, FEAT_DIM, FEAT_DIM, FEAT_DIM, FEAT_DIM, 0, 0, 0);
}
